// ConceptQuerySlotAttention_70171175682235
// MI455X (gfx1250) — matrix operand order repaired and checked by reading, not compiled, not run on hardware
//
#include <hip/hip_runtime.h>
#include <hip/hip_bf16.h>
#include <math.h>

// ---------------- constants (match reference) ----------------
#define BB 32
#define NN 1024
#define CC 64
#define DD 768
#define D3 (3 * DD)
#define EPSW 1e-8f
#define LNEPS 1e-5f

typedef __bf16 bf16_t;
typedef __attribute__((ext_vector_type(16))) __bf16 v16bf;
typedef __attribute__((ext_vector_type(8)))  __bf16 v8bf;
typedef __attribute__((ext_vector_type(8)))  float  v8f;
typedef __attribute__((ext_vector_type(4))) unsigned int u32x4;
typedef __attribute__((ext_vector_type(8))) int i32x8;
typedef __attribute__((ext_vector_type(4))) int i32x4;

// ---------------- WMMA fragment loaders (gfx1250 16x16x32 bf16 layouts) ----
// A (16x32, MxK): lane l holds row M=l&15; halves 0..7 = K in [koff, koff+8),
// halves 8..15 = K in [16+koff, 16+koff+8), koff = (l>=16)?8:0.
__device__ __forceinline__ v16bf load_a16(const bf16_t* __restrict__ A, int lda,
                                          int tm, int k, int lane) {
  const int row  = tm + (lane & 15);
  const int koff = (lane >> 4) << 3;
  const bf16_t* p = A + (size_t)row * lda + k + koff;
  v8bf x0 = *(const v8bf*)p;
  v8bf x1 = *(const v8bf*)(p + 16);
  v16bf f;
#pragma unroll
  for (int i = 0; i < 8; ++i) { f[i] = x0[i]; f[i + 8] = x1[i]; }
  return f;
}

// B (32x16, KxN): lane l holds column N=l&15; halves 0..7 = K in [koff, koff+8),
// halves 8..15 = K in [16+koff, 16+koff+8), koff = (l>=16)?8:0.
// B memory layout: [Ncols, K] with K contiguous (i.e. W[out,in]).
__device__ __forceinline__ v16bf load_b16(const bf16_t* __restrict__ Bm, int ldb,
                                          int tn, int k, int lane) {
  const int col = tn + (lane & 15);
  const bf16_t* p = Bm + (size_t)col * ldb + k + ((lane >> 4) << 3);
  v8bf x0 = *(const v8bf*)p;
  v8bf x1 = *(const v8bf*)(p + 16);
  v16bf f;
#pragma unroll
  for (int i = 0; i < 8; ++i) { f[i] = x0[i]; f[i + 8] = x1[i]; }
  return f;
}

// ---------------- generic bf16 WMMA GEMM:  Out[M,N] = A[M,K] * B[N,K]^T ----
// Wave tile: 32(M) x 64(N) -> 8 WMMAs per k-step sharing 4 B fragments.
// Block: 4 waves -> 128(M) x 64(N).
// MODE 0: store fp32 row-major.  MODE 1: store bf16 row-major.
// MODE 2: store bf16 transposed per 1024-row batch: Out[b][col][m%1024].
template <int MODE>
__global__ __launch_bounds__(128) void wmma_gemm(const bf16_t* __restrict__ A,
                                                 const bf16_t* __restrict__ Bw,
                                                 void* __restrict__ Out,
                                                 int M, int Nout, int K) {
  const int lane = threadIdx.x & 31;
  const int wave = threadIdx.x >> 5;
  const int tm = blockIdx.y * 128 + wave * 32;
  const int tn = blockIdx.x * 64;

  v8f acc[2][4];
#pragma unroll
  for (int s = 0; s < 2; ++s)
#pragma unroll
    for (int t = 0; t < 4; ++t)
#pragma unroll
      for (int i = 0; i < 8; ++i) acc[s][t][i] = 0.0f;

  for (int k = 0; k < K; k += 32) {
    v16bf a0 = load_a16(A, K, tm, k, lane);
    v16bf a1 = load_a16(A, K, tm + 16, k, lane);
#pragma unroll
    for (int t = 0; t < 4; ++t) {
      v16bf bf_ = load_b16(Bw, K, tn + t * 16, k, lane);
      acc[0][t] = __builtin_amdgcn_wmma_f32_16x16x32_bf16(
          false, a0, false, bf_, (short)0, acc[0][t], false, false);
      acc[1][t] = __builtin_amdgcn_wmma_f32_16x16x32_bf16(
          false, a1, false, bf_, (short)0, acc[1][t], false, false);
    }
  }

  // C/D layout: VGPR r -> (M = r + rb, N = lane&15), rb = (lane>=16)?8:0
  const int cl = lane & 15;
  const int rb = (lane >> 4) << 3;
#pragma unroll
  for (int s = 0; s < 2; ++s) {
    const int ms = tm + s * 16;
    if (MODE == 0) {
      float* O = (float*)Out;
#pragma unroll
      for (int t = 0; t < 4; ++t) {
        const int col = tn + t * 16 + cl;
#pragma unroll
        for (int r = 0; r < 8; ++r)
          O[(size_t)(ms + rb + r) * Nout + col] = acc[s][t][r];
      }
    } else if (MODE == 1) {
      bf16_t* O = (bf16_t*)Out;
#pragma unroll
      for (int t = 0; t < 4; ++t) {
        const int col = tn + t * 16 + cl;
#pragma unroll
        for (int r = 0; r < 8; ++r)
          O[(size_t)(ms + rb + r) * Nout + col] = (bf16_t)acc[s][t][r];
      }
    } else {  // transposed bf16 store; 8 contiguous M per lane -> one b128 store
      bf16_t* O = (bf16_t*)Out;
      const int mg  = ms + rb;
      const int b   = mg >> 10;
      const int nin = mg & 1023;
#pragma unroll
      for (int t = 0; t < 4; ++t) {
        const int col = tn + t * 16 + cl;
        v8bf v;
#pragma unroll
        for (int r = 0; r < 8; ++r) v[r] = (bf16_t)acc[s][t][r];
        *(v8bf*)(O + (size_t)b * Nout * 1024 + (size_t)col * 1024 + nin) = v;
      }
    }
  }
}

// ---------------- logits = (q @ k^T) * scale, masked-fill(mask==0 -> 0) -----
__global__ __launch_bounds__(128) void logits_kernel(const bf16_t* __restrict__ Q,
                                                     const bf16_t* __restrict__ Kb,
                                                     const int* __restrict__ mask,
                                                     float* __restrict__ L,
                                                     float scale) {
  const int b    = blockIdx.z;
  const int lane = threadIdx.x & 31;
  const int wave = threadIdx.x >> 5;
  const int tm = wave * 16;           // 64 slot rows per batch
  const int tn = blockIdx.x * 64;     // n columns
  const bf16_t* A  = Q  + (size_t)b * CC * DD;
  const bf16_t* Bm = Kb + (size_t)b * NN * DD;
  float* Ob = L + (size_t)b * CC * NN;

  v8f acc[4];
#pragma unroll
  for (int t = 0; t < 4; ++t)
#pragma unroll
    for (int i = 0; i < 8; ++i) acc[t][i] = 0.0f;

  for (int k = 0; k < DD; k += 32) {
    v16bf af = load_a16(A, DD, tm, k, lane);
#pragma unroll
    for (int t = 0; t < 4; ++t) {
      v16bf bf_ = load_b16(Bm, DD, tn + t * 16, k, lane);
      acc[t] = __builtin_amdgcn_wmma_f32_16x16x32_bf16(
          false, af, false, bf_, (short)0, acc[t], false, false);
    }
  }

  const int cl = lane & 15;
  const int rb = (lane >> 4) << 3;
#pragma unroll
  for (int t = 0; t < 4; ++t) {
    const int col = tn + t * 16 + cl;
    const int mv  = mask[b * NN + col];
#pragma unroll
    for (int r = 0; r < 8; ++r)
      Ob[(size_t)(tm + rb + r) * NN + col] = mv ? acc[t][r] * scale : 0.0f;
  }
}

// ---------------- updates = (attn/rowsum) @ v, A from fp32 + per-row scale --
__global__ __launch_bounds__(128) void updates_kernel(const float* __restrict__ attn,
                                                      const float* __restrict__ S,
                                                      const bf16_t* __restrict__ vT,
                                                      bf16_t* __restrict__ U) {
  const int b    = blockIdx.z;
  const int lane = threadIdx.x & 31;
  const int wave = threadIdx.x >> 5;
  const int tm = wave * 16;           // slot rows (C=64)
  const int tn = blockIdx.x * 64;     // d columns
  const float*  Ab = attn + (size_t)b * CC * NN;
  const bf16_t* Bb = vT   + (size_t)b * DD * NN;

  const int row  = tm + (lane & 15);
  const int koff = (lane >> 4) << 3;
  const float sA = 1.0f / (S[b * CC + row] + EPSW);

  v8f acc[4];
#pragma unroll
  for (int t = 0; t < 4; ++t)
#pragma unroll
    for (int i = 0; i < 8; ++i) acc[t][i] = 0.0f;

  for (int k = 0; k < NN; k += 32) {
    const float* p = Ab + (size_t)row * NN + k + koff;
    v8f x0 = *(const v8f*)p;
    v8f x1 = *(const v8f*)(p + 16);
    v16bf af;
#pragma unroll
    for (int i = 0; i < 8; ++i) {
      af[i]     = (bf16_t)(x0[i] * sA);
      af[i + 8] = (bf16_t)(x1[i] * sA);
    }
#pragma unroll
    for (int t = 0; t < 4; ++t) {
      v16bf bf_ = load_b16(Bb, NN, tn + t * 16, k, lane);
      acc[t] = __builtin_amdgcn_wmma_f32_16x16x32_bf16(
          false, af, false, bf_, (short)0, acc[t], false, false);
    }
  }

  const int cl = lane & 15;
  const int rb = (lane >> 4) << 3;
#pragma unroll
  for (int t = 0; t < 4; ++t) {
    const int col = tn + t * 16 + cl;
#pragma unroll
    for (int r = 0; r < 8; ++r)
      U[(size_t)(b * CC + tm + rb + r) * DD + col] = (bf16_t)acc[t][r];
  }
}

// ---------------- LayerNorm (fp32 in, bf16 out), one row per block ---------
__global__ __launch_bounds__(256) void layernorm_bf16_kernel(const float* __restrict__ X,
                                                             const float* __restrict__ g,
                                                             const float* __restrict__ bt,
                                                             bf16_t* __restrict__ Y) {
  __shared__ float red[256];
  const int row = blockIdx.x;
  const int tid = threadIdx.x;
  const float* x = X + (size_t)row * DD;

  float xv[3];
  float s = 0.0f;
#pragma unroll
  for (int i = 0; i < 3; ++i) { xv[i] = x[tid + i * 256]; s += xv[i]; }
  red[tid] = s;
  __syncthreads();
  for (int off = 128; off > 0; off >>= 1) {
    if (tid < off) red[tid] += red[tid + off];
    __syncthreads();
  }
  const float mu = red[0] / (float)DD;
  __syncthreads();
  float s2 = 0.0f;
#pragma unroll
  for (int i = 0; i < 3; ++i) { float d = xv[i] - mu; s2 += d * d; }
  red[tid] = s2;
  __syncthreads();
  for (int off = 128; off > 0; off >>= 1) {
    if (tid < off) red[tid] += red[tid + off];
    __syncthreads();
  }
  const float rstd = rsqrtf(red[0] / (float)DD + LNEPS);
#pragma unroll
  for (int i = 0; i < 3; ++i) {
    const int d = tid + i * 256;
    Y[(size_t)row * DD + d] = (bf16_t)((xv[i] - mu) * rstd * g[d] + bt[d]);
  }
}

// ---------------- normmax (alpha=5) bisection over the slot axis -----------
// LDS tile is filled by the Tensor Data Mover (2D tile: 64 rows x 128 cols fp32)
// z^(1.25) = z*sqrt(sqrt(z)); z^(0.25) = sqrt(sqrt(z))
__global__ __launch_bounds__(128) void normmax_kernel(const float* __restrict__ L,
                                                      const int* __restrict__ mask,
                                                      float* __restrict__ attn) {
  __shared__ float z[CC][128];
  const int b  = blockIdx.y;
  const int n0 = blockIdx.x * 128;
  const float* Lb = L + (size_t)b * CC * NN;

#if defined(__gfx1250__)
  if (threadIdx.x < 32) {
    // ---- TDM descriptor (D#): 2D tensor, tile 128x64, fp32, dense load ----
    const unsigned long long ga = (unsigned long long)(size_t)(Lb + n0);
    // generic shared-aperture address: low 32 bits == LDS byte offset
    const unsigned int la = (unsigned int)(size_t)(&z[0][0]);
    u32x4 g0;
    g0[0] = 1u;                                   // count=1, user D#, no gather
    g0[1] = la;                                   // lds_addr
    g0[2] = (unsigned int)ga;                     // global_addr[31:0]
    g0[3] = (unsigned int)(ga >> 32) | (2u << 30);// global_addr[56:32] | type=2
    i32x8 g1;
    g1[0] = 2 << 16;        // workgroup_mask=0 | data_size=2 (4B) | no flags
    g1[1] = NN << 16;       // atomic_barrier_addr=0 | tensor_dim0[15:0]=1024
    g1[2] = CC << 16;       // tensor_dim0[31:16]=0 | tensor_dim1[15:0]=64
    g1[3] = 128 << 16;      // tensor_dim1[31:16]=0 | tile_dim0=128
    g1[4] = CC;             // tile_dim1=64 | tile_dim2=0
    g1[5] = NN;             // tensor_dim0_stride[31:0]=1024
    g1[6] = 0;              // stride0[47:32]=0 | tensor_dim1_stride[15:0]=0
    g1[7] = 0;              // tensor_dim1_stride[47:16]=0
    i32x4 gz = {0, 0, 0, 0};
#if __clang_major__ >= 23
    i32x8 gz8 = {0, 0, 0, 0, 0, 0, 0, 0};
    __builtin_amdgcn_tensor_load_to_lds(g0, g1, gz, gz, gz8, 0);
#else
    __builtin_amdgcn_tensor_load_to_lds(g0, g1, gz, gz, 0);
#endif
    __builtin_amdgcn_s_wait_tensorcnt(0);
  }
  __syncthreads();
#else
  for (int idx = threadIdx.x; idx < CC * 128; idx += 128) {
    const int c = idx >> 7, nl = idx & 127;
    z[c][nl] = Lb[(size_t)c * NN + n0 + nl];
  }
  __syncthreads();
#endif

  const int nl = threadIdx.x;
  float mx = z[0][nl];
  for (int c = 1; c < CC; ++c) mx = fmaxf(mx, z[c][nl]);
  float lo = mx - 1.0f, hi = mx;
  for (int it = 0; it < 50; ++it) {
    const float tau = 0.5f * (lo + hi);
    float f = -1.0f;
    for (int c = 0; c < CC; ++c) {
      float u = fmaxf(z[c][nl] - tau, 0.0f);
      f += u * sqrtf(sqrtf(u));            // u^(alpha/(alpha-1)) = u^1.25
    }
    if (f >= 0.0f) lo = tau; else hi = tau;
  }
  const float tau = 0.5f * (lo + hi);
  float ssum = 0.0f;
  for (int c = 0; c < CC; ++c) {
    const float u = fmaxf(z[c][nl] - tau, 0.0f);
    ssum += (u > 0.0f) ? sqrtf(sqrtf(u)) : 0.0f;   // u^(1/(alpha-1)) = u^0.25
  }
  const float inv = 1.0f / ssum;
  const int mv = mask[b * NN + n0 + nl];
  float* Ab = attn + (size_t)b * CC * NN;
  for (int c = 0; c < CC; ++c) {
    const float u = fmaxf(z[c][nl] - tau, 0.0f);
    float p = (u > 0.0f) ? sqrtf(sqrtf(u)) * inv : 0.0f;
    Ab[(size_t)c * NN + n0 + nl] = mv ? p : 0.0f;
  }
}

// ---------------- deterministic row sums: S[row] = sum_n attn[row, n] ------
__global__ __launch_bounds__(256) void rowsum_kernel(const float* __restrict__ X,
                                                     float* __restrict__ S, int ld) {
  __shared__ float red[256];
  const int row = blockIdx.x;
  const float* x = X + (size_t)row * ld;
  float s = 0.0f;
  for (int i = threadIdx.x; i < ld; i += 256) s += x[i];
  red[threadIdx.x] = s;
  __syncthreads();
  for (int off = 128; off > 0; off >>= 1) {
    if (threadIdx.x < off) red[threadIdx.x] += red[threadIdx.x + off];
    __syncthreads();
  }
  if (threadIdx.x == 0) S[row] = red[0];
}

// ---------------- GRU cell elementwise (torch r,z,n gate order) ------------
__global__ void gru_kernel(const float* __restrict__ gi, const float* __restrict__ gh,
                           const float* __restrict__ b_ih, const float* __restrict__ b_hh,
                           const float* __restrict__ h0, float* __restrict__ out,
                           int total) {
  const int idx = blockIdx.x * blockDim.x + threadIdx.x;
  if (idx >= total) return;
  const int row = idx / DD, d = idx % DD;
  const float* gir = gi + (size_t)row * D3;
  const float* ghr = gh + (size_t)row * D3;
  const float ir = gir[d]           + b_ih[d],           hr = ghr[d]           + b_hh[d];
  const float iz = gir[DD + d]      + b_ih[DD + d],      hz = ghr[DD + d]      + b_hh[DD + d];
  const float in_ = gir[2 * DD + d] + b_ih[2 * DD + d],  hn = ghr[2 * DD + d]  + b_hh[2 * DD + d];
  const float r  = 1.0f / (1.0f + expf(-(ir + hr)));
  const float zg = 1.0f / (1.0f + expf(-(iz + hz)));
  const float ng = tanhf(in_ + r * hn);
  out[idx] = (1.0f - zg) * ng + zg * h0[idx];
}

// ---------------- small helpers --------------------------------------------
__global__ void f32_to_bf16_kernel(const float* __restrict__ in,
                                   bf16_t* __restrict__ out, int n) {
  const int i = blockIdx.x * blockDim.x + threadIdx.x;
  if (i < n) out[i] = (bf16_t)in[i];
}

// ---------------- host launcher --------------------------------------------
extern "C" void kernel_launch(void* const* d_in, const int* in_sizes, int n_in,
                              void* d_out, int out_size, void* d_ws, size_t ws_size,
                              hipStream_t stream) {
  const float* features   = (const float*)d_in[0];
  const float* slots_init = (const float*)d_in[1];
  const int*   mask       = (const int*)d_in[2];
  const float* ln_f_scale = (const float*)d_in[3];
  const float* ln_f_bias  = (const float*)d_in[4];
  const float* ln_s_scale = (const float*)d_in[5];
  const float* ln_s_bias  = (const float*)d_in[6];
  const float* Wk  = (const float*)d_in[7];
  const float* Wv  = (const float*)d_in[8];
  const float* Wq  = (const float*)d_in[9];
  const float* Wih = (const float*)d_in[10];
  const float* Whh = (const float*)d_in[11];
  const float* b_ih = (const float*)d_in[12];
  const float* b_hh = (const float*)d_in[13];

  float* out_slots = (float*)d_out;                         // [B,C,D]
  float* out_attn  = out_slots + (size_t)BB * CC * DD;      // [B,C,N]

  // workspace carve-out (256B aligned regions)
  char* ws = (char*)d_ws;
  size_t off = 0;
  auto alloc = [&](size_t bytes) -> void* {
    void* p = (void*)(ws + off);
    off = (off + bytes + 255) & ~(size_t)255;
    return p;
  };
  bf16_t* Wk_bf  = (bf16_t*)alloc((size_t)DD * DD * 2);
  bf16_t* Wv_bf  = (bf16_t*)alloc((size_t)DD * DD * 2);
  bf16_t* Wq_bf  = (bf16_t*)alloc((size_t)DD * DD * 2);
  bf16_t* Wih_bf = (bf16_t*)alloc((size_t)D3 * DD * 2);
  bf16_t* Whh_bf = (bf16_t*)alloc((size_t)D3 * DD * 2);
  bf16_t* h0_bf  = (bf16_t*)alloc((size_t)BB * CC * DD * 2);
  bf16_t* f_bf   = (bf16_t*)alloc((size_t)BB * NN * DD * 2);
  bf16_t* s_bf   = (bf16_t*)alloc((size_t)BB * CC * DD * 2);
  bf16_t* k_bf   = (bf16_t*)alloc((size_t)BB * NN * DD * 2);
  bf16_t* vT_bf  = (bf16_t*)alloc((size_t)BB * DD * NN * 2);
  bf16_t* q_bf   = (bf16_t*)alloc((size_t)BB * CC * DD * 2);
  float*  logitsF = (float*)alloc((size_t)BB * CC * NN * 4);
  float*  Ssum    = (float*)alloc((size_t)BB * CC * 4);
  bf16_t* upd_bf  = (bf16_t*)alloc((size_t)BB * CC * DD * 2);
  float*  giF     = (float*)alloc((size_t)BB * CC * D3 * 4);
  float*  ghF     = (float*)alloc((size_t)BB * CC * D3 * 4);
  (void)ws_size; (void)n_in; (void)in_sizes; (void)out_size;

  const dim3 blk128(128);
  const float scale = 1.0f / sqrtf((float)DD);

  // weight / h0 bf16 conversion
  auto cvt = [&](const float* src, bf16_t* dst, int n) {
    f32_to_bf16_kernel<<<(n + 255) / 256, 256, 0, stream>>>(src, dst, n);
  };
  cvt(Wk,  Wk_bf,  DD * DD);
  cvt(Wv,  Wv_bf,  DD * DD);
  cvt(Wq,  Wq_bf,  DD * DD);
  cvt(Wih, Wih_bf, D3 * DD);
  cvt(Whh, Whh_bf, D3 * DD);
  cvt(slots_init, h0_bf, BB * CC * DD);

  // LayerNorms
  layernorm_bf16_kernel<<<BB * NN, 256, 0, stream>>>(features, ln_f_scale, ln_f_bias, f_bf);
  layernorm_bf16_kernel<<<BB * CC, 256, 0, stream>>>(slots_init, ln_s_scale, ln_s_bias, s_bf);

  // k = f @ Wk^T (bf16, row-major) ; v = f @ Wv^T (bf16, transposed per batch)
  wmma_gemm<1><<<dim3(DD / 64, (BB * NN) / 128), blk128, 0, stream>>>(f_bf, Wk_bf, k_bf, BB * NN, DD, DD);
  wmma_gemm<2><<<dim3(DD / 64, (BB * NN) / 128), blk128, 0, stream>>>(f_bf, Wv_bf, vT_bf, BB * NN, DD, DD);
  // q = ln(slots) @ Wq^T
  wmma_gemm<1><<<dim3(DD / 64, (BB * CC) / 128), blk128, 0, stream>>>(s_bf, Wq_bf, q_bf, BB * CC, DD, DD);

  // logits (masked, scaled, fp32)
  logits_kernel<<<dim3(NN / 64, 1, BB), blk128, 0, stream>>>(q_bf, k_bf, mask, logitsF, scale);

  // normmax over slot axis -> attn output; then deterministic row sums
  normmax_kernel<<<dim3(NN / 128, BB), blk128, 0, stream>>>(logitsF, mask, out_attn);
  rowsum_kernel<<<BB * CC, 256, 0, stream>>>(out_attn, Ssum, NN);

  // updates = (attn / (rowsum+eps)) @ v
  updates_kernel<<<dim3(DD / 64, 1, BB), blk128, 0, stream>>>(out_attn, Ssum, vT_bf, upd_bf);

  // GRU GEMMs (fp32 out) + elementwise gates
  wmma_gemm<0><<<dim3(D3 / 64, (BB * CC) / 128), blk128, 0, stream>>>(upd_bf, Wih_bf, giF, BB * CC, D3, DD);
  wmma_gemm<0><<<dim3(D3 / 64, (BB * CC) / 128), blk128, 0, stream>>>(h0_bf, Whh_bf, ghF, BB * CC, D3, DD);
  gru_kernel<<<((BB * CC * DD) + 255) / 256, 256, 0, stream>>>(giF, ghF, b_ih, b_hh, slots_init, out_slots, BB * CC * DD);
}
